// GraphConv_6098853560487
// MI455X (gfx1250) — compile-verified, matrix operand order checked by reading, not run on hardware
//
#include <hip/hip_runtime.h>

// ---------------------------------------------------------------------------
// Fused e3nn-style GraphConv for MI455X (gfx1250, wave32, WMMA).
// GEMM1 computed transposed (h^T = w1^T x es^T) so its WMMA C-registers are
// already in GEMM2 A-fragment layout -> h never touches LDS.
// ---------------------------------------------------------------------------

typedef __attribute__((ext_vector_type(16))) _Float16 v16h;
typedef __attribute__((ext_vector_type(8)))  _Float16 v8h;
typedef __attribute__((ext_vector_type(8)))  float    v8f;
typedef __attribute__((ext_vector_type(4)))  float    v4f;

#define N_NODES   10000
#define N_EDGES   160000
#define DIM_IN    40        // 16 (0e) + 8*3 (1o)
#define N_BASIS   16
#define HIDDEN    256
#define WNUMEL    576       // 256 + 128 + 64 + 128
#define QT1       16        // HIDDEN/16  M-tiles of GEMM1'
#define NT2       36        // WNUMEL/16  N-tiles of GEMM2
#define KK2       8         // HIDDEN/32  K-steps of GEMM2
#define WAVES     4
#define EPW       16        // edges per wave (one WMMA M/N tile)
#define EPB       (WAVES * EPW)          // 64 edges/block -> 2500 blocks
#define WPAD      20        // padded wtile row (floats): 80B, 16B-aligned rows

#define A0        0.20412414523193154f   // 1/sqrt(24)  (== a1*INV_SQRT3)
#define A0INV3    0.11785113019775793f   // 1/sqrt(72)
#define RELU_GAIN 1.4142135623730951f

// ---------------------------------------------------------------------------
// Prep: zero output; build w1 in A-fragment order (transposed, /sqrt(16)) and
// w2 in B-fragment order (/sqrt(256)), both f16.
// 16-bit fragment K mapping: element e of lane l holds
//   K = (e<8 ? e : e+8) + (l>=16 ? 8 : 0)  (+ 32*kk for GEMM2)
// ---------------------------------------------------------------------------
__global__ void prep_kernel(const float* __restrict__ w1,
                            const float* __restrict__ w2,
                            _Float16* __restrict__ w1fA,
                            _Float16* __restrict__ w2f,
                            float* __restrict__ out, int total) {
  int i = blockIdx.x * blockDim.x + threadIdx.x;
  if (i >= total) return;
  const int OUT_N = N_NODES * DIM_IN;
  const int W1F   = QT1 * 32 * 16;
  if (i < OUT_N) { out[i] = 0.0f; return; }
  i -= OUT_N;
  if (i < W1F) {            // A-frag: lane's M = hidden = q*16 + (lane%16)
    int e = i & 15, lane = (i >> 4) & 31, q = i >> 9;
    int m = lane & 15;
    int K = (e < 8 ? e : e + 8) + ((lane >= 16) ? 8 : 0);
    float v = (K < N_BASIS) ? w1[K * HIDDEN + q * 16 + m] * 0.25f : 0.0f;
    w1fA[i] = (_Float16)v;
    return;
  }
  i -= W1F;
  {                         // B-frag: lane's N = q*16 + (lane%16), K = hidden
    int e = i & 15, lane = (i >> 4) & 31, tk = i >> 9;
    int kk = tk & 7, t = tk >> 3;
    int n = lane & 15;
    int K = kk * 32 + (e < 8 ? e : e + 8) + ((lane >= 16) ? 8 : 0);
    w2f[i] = (_Float16)(w2[K * WNUMEL + t * 16 + n] * 0.0625f);
  }
}

// ---------------------------------------------------------------------------
// Main fused kernel: 1 wave = 16 edges; 4 waves/block share w2f via L1
// (per-tile barrier keeps them in lock-step on the same B fragments).
// ---------------------------------------------------------------------------
__global__ __launch_bounds__(WAVES * 32)
void graphconv_kernel(const float* __restrict__ nf,
                      const float* __restrict__ ea,
                      const float* __restrict__ es,
                      const _Float16* __restrict__ w1fA,
                      const _Float16* __restrict__ w2f,
                      const int* __restrict__ src,
                      const int* __restrict__ dst,
                      const int* __restrict__ nn,
                      float* __restrict__ out) {
  __shared__ float wtile[WAVES][2][16][WPAD];   // weight-tile lane transpose

  const int lane  = threadIdx.x & 31;
  const int wave  = threadIdx.x >> 5;
  const int m     = lane & 15;
  const int hi    = lane >> 4;
  const int eBase = (blockIdx.x * WAVES + wave) * EPW;
  const int e     = eBase + m;                  // lane pair (m, m+16) <-> edge

  // -------- GEMM1' : h^T = (w1^T/sqrt16) x es^T, output stays in VGPRs ----
  v16h besf;                                    // es B-fragment (K=16 padded)
  {
    const v8f sv = *(const v8f*)(es + (size_t)e * N_BASIS + hi * 8);
#pragma unroll
    for (int k = 0; k < 8; ++k) besf[k] = (_Float16)sv[k];
#pragma unroll
    for (int k = 8; k < 16; ++k) besf[k] = (_Float16)0.0f;
  }
  v8h hq[QT1];
#pragma unroll
  for (int q = 0; q < QT1; ++q) {
    v16h afr = *(const v16h*)(w1fA + ((q * 32 + lane) << 4));
    v8f c = {};
    c = __builtin_amdgcn_wmma_f32_16x16x32_f16(false, afr, false, besf,
                                               (short)0, c, false, false);
    v8h hh;
#pragma unroll
    for (int v = 0; v < 8; ++v)
      hh[v] = (_Float16)(fmaxf(c[v], 0.0f) * RELU_GAIN);
    hq[q] = hh;
  }
  // C-layout of tiles (2kk, 2kk+1) == A-fragment for K-step kk of GEMM2.
  v16h af[KK2];
#pragma unroll
  for (int kk = 0; kk < KK2; ++kk)
    af[kk] = __builtin_shufflevector(hq[2 * kk], hq[2 * kk + 1],
        0,1,2,3,4,5,6,7,8,9,10,11,12,13,14,15);

  // -------- per-edge tensor-product coefficients --------------------------
  const float* nrow = nf + (size_t)src[e] * DIM_IN;
  const float y0  = ea[(size_t)e * 4 + 0];
  const float y1a = ea[(size_t)e * 4 + 1];
  const float y1b = ea[(size_t)e * 4 + 2];
  const float y1c = ea[(size_t)e * 4 + 3];
  float cA[16], t1[8], cB0[8], cB1[8], cB2[8];
#pragma unroll
  for (int u = 0; u < 16; ++u) cA[u] = A0 * nrow[u];
#pragma unroll
  for (int u = 0; u < 8; ++u) {
    float xa = nrow[16 + u * 3 + 0];
    float xb = nrow[16 + u * 3 + 1];
    float xc = nrow[16 + u * 3 + 2];
    t1[u]  = A0INV3 * (xa * y1a + xb * y1b + xc * y1c);
    cB0[u] = A0 * y0 * xa;
    cB1[u] = A0 * y0 * xb;
    cB2[u] = A0 * y0 * xc;
  }
  // low lane of pair accumulates out[0:16], high lane out[16:40]
  float acc[24];
#pragma unroll
  for (int i = 0; i < 24; ++i) acc[i] = 0.0f;

  // -------- GEMM2 (per-edge weights) fused with tensor product ------------
#pragma unroll
  for (int t = 0; t < NT2; ++t) {
    const int buf = t & 1;
    v8f c = {};
#pragma unroll
    for (int kk = 0; kk < KK2; ++kk) {
      v16h bfr = *(const v16h*)(w2f + (((t * KK2 + kk) * 32 + lane) << 4));
      c = __builtin_amdgcn_wmma_f32_16x16x32_f16(false, af[kk], false, bfr,
                                                 (short)0, c, false, false);
    }
    // stage C tile (rows = edges) for the lane transpose
#pragma unroll
    for (int v = 0; v < 8; ++v) wtile[wave][buf][hi * 8 + v][m] = c[v];
    __syncthreads();
    float wv[16];
#pragma unroll
    for (int jj = 0; jj < 4; ++jj) {
      v4f wq = *(const v4f*)&wtile[wave][buf][m][jj * 4];
      wv[jj * 4 + 0] = wq[0]; wv[jj * 4 + 1] = wq[1];
      wv[jj * 4 + 2] = wq[2]; wv[jj * 4 + 3] = wq[3];
    }
    if (t < 16) {                       // w00: u == t, w = j  (out0, low lane)
      if (hi == 0) {
        const float su = cA[t] * y0;
#pragma unroll
        for (int j = 0; j < 16; ++j) acc[j] += su * wv[j];
      }
    } else if (t < 24) {                // w01: u=(t-16)*2+j/8, w=j%8 (high lane)
      if (hi == 1) {
#pragma unroll
        for (int j = 0; j < 16; ++j) {
          const int u = (t - 16) * 2 + (j >> 3);
          const int w = j & 7;
          const float base = cA[u] * wv[j];
          acc[3 * w + 0] += base * y1a;
          acc[3 * w + 1] += base * y1b;
          acc[3 * w + 2] += base * y1c;
        }
      }
    } else if (t < 28) {                // w10: u=(t-24)*2+j/8, w=j%8 (high lane)
      if (hi == 1) {
#pragma unroll
        for (int j = 0; j < 16; ++j) {
          const int u = (t - 24) * 2 + (j >> 3);
          const int w = j & 7;
          acc[3 * w + 0] += cB0[u] * wv[j];
          acc[3 * w + 1] += cB1[u] * wv[j];
          acc[3 * w + 2] += cB2[u] * wv[j];
        }
      }
    } else {                            // w110: u = t-28, w = j (out0, low lane)
      if (hi == 0) {
        const float tu = t1[t - 28];
#pragma unroll
        for (int j = 0; j < 16; ++j) acc[j] += tu * wv[j];
      }
    }
  }

  // -------- scatter-add (lane pair splits the 40 outputs) -----------------
  const float scale = rsqrtf((float)nn[0]);
  float* orow = out + (size_t)dst[e] * DIM_IN;
  if (hi == 0) {
#pragma unroll
    for (int i = 0; i < 16; ++i) atomicAdd(orow + i, acc[i] * scale);
  } else {
#pragma unroll
    for (int i = 0; i < 24; ++i) atomicAdd(orow + 16 + i, acc[i] * scale);
  }
}

// ---------------------------------------------------------------------------
extern "C" void kernel_launch(void* const* d_in, const int* in_sizes, int n_in,
                              void* d_out, int out_size, void* d_ws, size_t ws_size,
                              hipStream_t stream) {
  const float* nf  = (const float*)d_in[0];   // node_features (10000, 40)
  const float* ea  = (const float*)d_in[1];   // edge_attr     (160000, 4)
  const float* es  = (const float*)d_in[2];   // edge_scalars  (160000, 16)
  const float* w1  = (const float*)d_in[3];   // (16, 256)
  const float* w2  = (const float*)d_in[4];   // (256, 576)
  const int*   src = (const int*)d_in[5];
  const int*   dst = (const int*)d_in[6];
  const int*   nn  = (const int*)d_in[7];     // scalar num_neighbors
  float* out = (float*)d_out;

  _Float16* w1fA = (_Float16*)d_ws;                      // 16*32*16   = 8192 halfs
  _Float16* w2f  = w1fA + (QT1 * 32 * 16);               // 36*8*32*16 = 147456 halfs

  const int prep_total = N_NODES * DIM_IN + QT1 * 32 * 16 + NT2 * KK2 * 32 * 16;
  prep_kernel<<<(prep_total + 255) / 256, 256, 0, stream>>>(w1, w2, w1fA, w2f,
                                                            out, prep_total);

  const int blocks = N_EDGES / EPB;                      // 2500 exactly
  graphconv_kernel<<<blocks, WAVES * 32, 0, stream>>>(nf, ea, es, w1fA, w2f,
                                                      src, dst, nn, out);
}
